// MultiHeadAttention_30803505447235
// MI455X (gfx1250) — matrix operand order repaired and checked by reading, not compiled, not run on hardware
//
#include <hip/hip_runtime.h>
#include <hip/hip_bf16.h>

#define D_MODEL 1024
#define SEQ     2048
#define NHEAD   16
#define DKH     64
#define NBATCH  4

typedef __attribute__((ext_vector_type(16))) __bf16 v16bf;
typedef __attribute__((ext_vector_type(8)))  __bf16 v8bf;
typedef __attribute__((ext_vector_type(4)))  __bf16 v4bf;
typedef __attribute__((ext_vector_type(8)))  float  v8f;

__device__ __forceinline__ v8f wmma_bf16(v16bf a, v16bf b, v8f c) {
  // 8 args: (neg_a, A, neg_b, B, c_mod, C, reuse_a, reuse_b)
  return __builtin_amdgcn_wmma_f32_16x16x32_bf16(false, a, false, b, (short)0, c,
                                                 false, false);
}

// A-fragment (16x32 bf16): lane L (M=L%16, half=L/16) holds
//   elems e -> K = kbase + half*16 + e
__device__ __forceinline__ v16bf a_frag_f32(const float* row, int kbase, int half) {
  v8f x0 = *(const v8f*)(row + kbase + half * 16);
  v8f x1 = *(const v8f*)(row + kbase + half * 16 + 8);
  v16bf r;
#pragma unroll
  for (int i = 0; i < 8; ++i) { r[i] = (__bf16)x0[i]; r[i + 8] = (__bf16)x1[i]; }
  return r;
}
__device__ __forceinline__ v16bf a_frag_bf16(const __bf16* row, int kbase, int half) {
  v8bf x0 = *(const v8bf*)(row + kbase + half * 16);
  v8bf x1 = *(const v8bf*)(row + kbase + half * 16 + 8);
  v16bf r;
#pragma unroll
  for (int i = 0; i < 8; ++i) { r[i] = x0[i]; r[i + 8] = x1[i]; }
  return r;
}
// B-fragment (32x16 bf16): lane L (N=L%16, half) holds elems e -> K = kbase + half*16 + e
__device__ __forceinline__ v16bf b_frag_f32(const float* col, int kbase, int half) {
  v8f x0 = *(const v8f*)(col + kbase + half * 16);
  v8f x1 = *(const v8f*)(col + kbase + half * 16 + 8);
  v16bf r;
#pragma unroll
  for (int i = 0; i < 8; ++i) { r[i] = (__bf16)x0[i]; r[i + 8] = (__bf16)x1[i]; }
  return r;
}
__device__ __forceinline__ v16bf b_frag_bf16(const __bf16* col, int kbase, int half) {
  return *(const v16bf*)(col + kbase + half * 16);
}

// ---- 16-lane row-max reduction: pure-VALU v_permlane16_b32 butterflies ----
#if __has_builtin(__builtin_amdgcn_permlane16)
__device__ __forceinline__ float xor16(float v, unsigned s0, unsigned s1) {
  unsigned u = __builtin_bit_cast(unsigned, v);
  u = __builtin_amdgcn_permlane16(u, u, s0, s1, false, false);
  return __builtin_bit_cast(float, u);
}
__device__ __forceinline__ float rmax16(float v) {
  v = fmaxf(v, xor16(v, 0x67452301u, 0xEFCDAB89u)); // lane ^= 1
  v = fmaxf(v, xor16(v, 0x54761032u, 0xDCFE98BAu)); // lane ^= 2
  v = fmaxf(v, xor16(v, 0x32107654u, 0xBA98FEDCu)); // lane ^= 4
  v = fmaxf(v, xor16(v, 0xFEDCBA98u, 0x76543210u)); // lane ^= 8
  return v;
}
#else
__device__ __forceinline__ float rmax16(float v) {
  v = fmaxf(v, __shfl_xor(v, 1, 32));
  v = fmaxf(v, __shfl_xor(v, 2, 32));
  v = fmaxf(v, __shfl_xor(v, 4, 32));
  v = fmaxf(v, __shfl_xor(v, 8, 32));
  return v;
}
#endif

// Wave-private LDS round-trip ordering: same-wave DS ops are in-order in HW;
// this is a compiler memory barrier + DS-counter drain (no block barrier needed).
__device__ __forceinline__ void wave_lds_sync() {
  asm volatile("s_wait_dscnt 0x0" ::: "memory");
}

// out[m,n] = sum_k A[m,k] * W[n,k] + bias[n]
// OUT_MODE: 0 = bf16 row-major [M][N]; 1 = bf16 V-transposed+key-permuted; 2 = f32 row-major
template <int A_IS_BF16, int OUT_MODE>
__global__ __launch_bounds__(256) void proj_gemm(const void* __restrict__ Ain,
                                                 const float* __restrict__ W,
                                                 const float* __restrict__ bias,
                                                 void* __restrict__ Out) {
  const int lane = threadIdx.x & 31;
  const int wave = threadIdx.x >> 5;
  const int half = lane >> 4;
  const int l16  = lane & 15;
  const int n0 = blockIdx.x * 128 + wave * 16;
  const int m0 = blockIdx.y * 16;

  const float*  af = (const float*)Ain  + (size_t)(m0 + l16) * D_MODEL;
  const __bf16* ab = (const __bf16*)Ain + (size_t)(m0 + l16) * D_MODEL;
  const float*  wr = W + (size_t)(n0 + l16) * D_MODEL;

  v8f acc = {};
#pragma unroll 4
  for (int k0 = 0; k0 < D_MODEL; k0 += 32) {
    v16bf a = A_IS_BF16 ? a_frag_bf16(ab, k0, half) : a_frag_f32(af, k0, half);
    v16bf b = b_frag_f32(wr, k0, half);
    acc = wmma_bf16(a, b, acc);
  }
  const float bi = bias[n0 + l16];
#pragma unroll
  for (int r = 0; r < 8; ++r) {
    const int m = m0 + r + half * 8;   // C/D layout: VGPR r -> row r (lanes 0-15), r+8 (16-31)
    const int n = n0 + l16;
    const float val = acc[r] + bi;
    if (OUT_MODE == 2) {
      ((float*)Out)[(size_t)m * D_MODEL + n] = val;
    } else if (OUT_MODE == 0) {
      ((__bf16*)Out)[(size_t)m * D_MODEL + n] = (__bf16)val;
    } else {
      // [b][h][dk][seq'], seq permuted within each 64-block: pos = (s%16)*4 + s/16.
      // Must match the P-staging key order in attn_fa.
      const int h = n >> 6, dk = n & 63, bb = m >> 11, s = m & 2047;
      const int sin = s & 63;
      const int sp  = (s & ~63) | (((sin & 15) << 2) | (sin >> 4));
      ((__bf16*)Out)[(((size_t)(bb * NHEAD + h) * DKH) + dk) * SEQ + sp] = (__bf16)val;
    }
  }
}

// Flash-attention: one wave per 16-query block of one (b,h); stream 64 keys/step.
__global__ __launch_bounds__(128) void attn_fa(const __bf16* __restrict__ qp,
                                               const __bf16* __restrict__ kp,
                                               const __bf16* __restrict__ vpT,
                                               __bf16* __restrict__ ctx) {
  __shared__ __align__(64) __bf16 Ps[4][16][64];  // per-wave P staging, key-permuted order
  const int lane = threadIdx.x & 31;
  const int wave = threadIdx.x >> 5;
  const int half = lane >> 4;
  const int l16  = lane & 15;
  const int bh = blockIdx.x;
  const int b  = bh >> 4;
  const int h  = bh & 15;
  const int q0 = (blockIdx.y * 4 + wave) * 16;

  const __bf16* qrow = qp + ((size_t)b * SEQ + q0 + l16) * D_MODEL + h * DKH;
  const v16bf qa0 = a_frag_bf16(qrow, 0, half);    // dk 0..31
  const v16bf qa1 = a_frag_bf16(qrow, 32, half);   // dk 32..63

  const __bf16* kbase = kp + (size_t)b * SEQ * D_MODEL + h * DKH;
  const __bf16* vbase = vpT + (size_t)(b * NHEAD + h) * DKH * SEQ;

  v16bf ones;
#pragma unroll
  for (int i = 0; i < 16; ++i) ones[i] = (__bf16)1.0f;

  // exp2 domain: scores pre-scaled by (1/sqrt(64)) * log2(e)
  const float SCL = 0.125f * 1.44269504088896340736f;

  float mr[8], lr[8], alpha[8];
#pragma unroll
  for (int r = 0; r < 8; ++r) { mr[r] = -1e30f; lr[r] = 0.f; }
  v8f O[4] = {{}, {}, {}, {}};

  for (int key0 = 0; key0 < SEQ; key0 += 64) {
    // prefetch next slab (speculative; one 128B line per row segment)
#pragma unroll
    for (int t = 0; t < 4; ++t) {
      __builtin_prefetch(kbase + (size_t)(key0 + 64 + t * 16 + l16) * D_MODEL, 0, 0);
      __builtin_prefetch(vbase + (size_t)(t * 16 + l16) * SEQ + key0 + 64, 0, 0);
    }

    // ---- S = Q (16x64) x K^T over 4 key tiles ----
    v8f s[4];
#pragma unroll
    for (int t = 0; t < 4; ++t) {
      const __bf16* kr = kbase + (size_t)(key0 + t * 16 + l16) * D_MODEL;
      v8f z = {};
      z = wmma_bf16(qa0, b_frag_bf16(kr, 0, half), z);
      s[t] = wmma_bf16(qa1, b_frag_bf16(kr, 32, half), z);
    }

    // ---- online softmax (exp2 domain), row max via VALU permlane butterflies ----
#pragma unroll
    for (int r = 0; r < 8; ++r) {
      float a0 = s[0][r] * SCL;
      float a1 = s[1][r] * SCL;
      float a2 = s[2][r] * SCL;
      float a3 = s[3][r] * SCL;
      const float mx = rmax16(fmaxf(fmaxf(a0, a1), fmaxf(a2, a3)));
      const float mn = fmaxf(mr[r], mx);
      alpha[r] = __builtin_amdgcn_exp2f(mr[r] - mn);
      mr[r] = mn;
      v4bf pv;
      pv[0] = (__bf16)__builtin_amdgcn_exp2f(a0 - mn);
      pv[1] = (__bf16)__builtin_amdgcn_exp2f(a1 - mn);
      pv[2] = (__bf16)__builtin_amdgcn_exp2f(a2 - mn);
      pv[3] = (__bf16)__builtin_amdgcn_exp2f(a3 - mn);
      // packed b64 store; key order within slab = pos = col*4 + tile (matches vpT)
      *(v4bf*)&Ps[wave][r + half * 8][l16 * 4] = pv;
      O[0][r] *= alpha[r]; O[1][r] *= alpha[r];
      O[2][r] *= alpha[r]; O[3][r] *= alpha[r];
    }
    wave_lds_sync();
    const v16bf pf0 = a_frag_bf16(&Ps[wave][l16][0], 0, half);   // P pos 0..31
    const v16bf pf1 = a_frag_bf16(&Ps[wave][l16][0], 32, half);  // P pos 32..63
    wave_lds_sync();

    // ---- row sums on the matrix pipe: ls = P x ones(64x16); every col = rowsum ----
    v8f ls = {};
    ls = wmma_bf16(pf0, ones, ls);
    ls = wmma_bf16(pf1, ones, ls);
#pragma unroll
    for (int r = 0; r < 8; ++r) lr[r] = lr[r] * alpha[r] + ls[r];

    // ---- O += P (16x64) x V (64x64): V rows stored in same permuted key order ----
#pragma unroll
    for (int j = 0; j < 4; ++j) {
      const __bf16* vr = vbase + (size_t)(j * 16 + l16) * SEQ + key0;
      O[j] = wmma_bf16(pf0, b_frag_bf16(vr, 0, half), O[j]);
      O[j] = wmma_bf16(pf1, b_frag_bf16(vr, 32, half), O[j]);
    }
  }

#pragma unroll
  for (int r = 0; r < 8; ++r) {
    const float inv = 1.0f / lr[r];
    __bf16* orow = ctx + ((size_t)b * SEQ + q0 + r + half * 8) * D_MODEL + h * DKH;
    orow[l16]      = (__bf16)(O[0][r] * inv);
    orow[16 + l16] = (__bf16)(O[1][r] * inv);
    orow[32 + l16] = (__bf16)(O[2][r] * inv);
    orow[48 + l16] = (__bf16)(O[3][r] * inv);
  }
}

extern "C" void kernel_launch(void* const* d_in, const int* in_sizes, int n_in,
                              void* d_out, int out_size, void* d_ws, size_t ws_size,
                              hipStream_t stream) {
  (void)in_sizes; (void)n_in; (void)out_size; (void)ws_size;
  const float* q   = (const float*)d_in[0];
  const float* k   = (const float*)d_in[1];
  const float* v   = (const float*)d_in[2];
  const float* w_q = (const float*)d_in[3];
  const float* b_q = (const float*)d_in[4];
  const float* w_k = (const float*)d_in[5];
  const float* b_k = (const float*)d_in[6];
  const float* w_v = (const float*)d_in[7];
  const float* b_v = (const float*)d_in[8];
  const float* w_o = (const float*)d_in[9];
  const float* b_o = (const float*)d_in[10];

  char* ws = (char*)d_ws;
  const size_t SZ = (size_t)NBATCH * SEQ * D_MODEL * sizeof(__bf16); // 16 MiB each
  __bf16* qp  = (__bf16*)(ws + 0 * SZ);   // [b][s][h*64+dk] bf16
  __bf16* kpp = (__bf16*)(ws + 1 * SZ);   // [b][s][h*64+dk] bf16
  __bf16* vpT = (__bf16*)(ws + 2 * SZ);   // [b][h][dk][seq'] bf16 (transposed, key-permuted)
  __bf16* ctx = (__bf16*)(ws + 3 * SZ);   // [b][s][h*64+dk] bf16

  dim3 gproj(D_MODEL / 128, (NBATCH * SEQ) / 16);
  proj_gemm<0, 0><<<gproj, 256, 0, stream>>>(q, w_q, b_q, qp);
  proj_gemm<0, 0><<<gproj, 256, 0, stream>>>(k, w_k, b_k, kpp);
  proj_gemm<0, 1><<<gproj, 256, 0, stream>>>(v, w_v, b_v, vpT);
  attn_fa<<<dim3(NBATCH * NHEAD, SEQ / 16 / 4), 128, 0, stream>>>(qp, kpp, vpT, ctx);
  proj_gemm<1, 2><<<gproj, 256, 0, stream>>>(ctx, w_o, b_o, d_out);
}
